// Glm4MoeLiteAttentionMLA_206158430291
// MI455X (gfx1250) — matrix operand order repaired and checked by reading, compiled, not run on hardware
//
#include <hip/hip_runtime.h>
#include <hip/hip_bf16.h>
#include <math.h>

typedef __attribute__((ext_vector_type(16))) __bf16        v16bf;
typedef __attribute__((ext_vector_type(8)))  float         v8f;
typedef __attribute__((ext_vector_type(8)))  unsigned int  v8u;
typedef __attribute__((ext_vector_type(4)))  unsigned int  v4u;
typedef __attribute__((ext_vector_type(2)))  float         v2f;

#define S_LEN   2048
#define HIDDEN  2048
#define Q_LORA  1536
#define KV_LORA 512
#define NOPE    128
#define ROPE    64
#define V_DIM   128
#define NH      16
#define QKD     192   // NOPE + ROPE
#define KV_OUT  256   // NOPE + V_DIM

union frag_u { v8u u8; v4u u4[2]; };

__device__ __forceinline__ unsigned short f2bf(float f) {
  unsigned int u = __float_as_uint(f);
  u = u + 0x7FFFu + ((u >> 16) & 1u);   // round-to-nearest-even
  return (unsigned short)(u >> 16);
}

__device__ __forceinline__ unsigned int f2bf2(float lo, float hi) {
#if __has_builtin(__builtin_amdgcn_cvt_pk_bf16_f32)
  auto r = __builtin_amdgcn_cvt_pk_bf16_f32(lo, hi);
  return __builtin_bit_cast(unsigned int, r);
#else
  return (unsigned int)f2bf(lo) | ((unsigned int)f2bf(hi) << 16);
#endif
}

__device__ __forceinline__ v8f vzero8() {
  v8f z;
#pragma unroll
  for (int i = 0; i < 8; ++i) z[i] = 0.0f;
  return z;
}

__device__ __forceinline__ v8f wmma_bf16(v16bf a, v16bf b, v8f c) {
  return __builtin_amdgcn_wmma_f32_16x16x32_bf16(false, a, false, b, (short)0, c,
                                                 false, false);
}

// ---------------------------------------------------------------------------
// Generic GEMM: C(f32, MxN) = A(f32, MxK) * B(f32, KxN), bf16 WMMA math.
// BM=128, BN=64, BK=64. 256 threads = 8 waves (4x2 grid), wave tile 32x32.
// A staged row-major; B staged TRANSPOSED (Bt[n][k]) so every fragment is
// loaded with ds_load_b128. Requires M%128==0, N%64==0, K%64==0.
// ---------------------------------------------------------------------------
#define BM 128
#define BN 64
#define BK 64

__global__ __launch_bounds__(256) void gemm_bf16_kernel(
    const float* __restrict__ A, const float* __restrict__ B,
    float* __restrict__ C, int M, int N, int K) {
  __shared__ unsigned short As[BM][BK + 8];   // stride 144 B (16-aligned)
  __shared__ unsigned short Bt[BN][BK + 8];   // transposed, stride 144 B
  const int tid  = threadIdx.x;
  const int lane = tid & 31, wave = tid >> 5;
  const int wm = wave >> 1, wn = wave & 1;    // 4x2 wave grid
  const int mlo = lane & 15, hi = lane >> 4;
  const int kb8 = hi * 8;
  const long bm = (long)blockIdx.y * BM;
  const long bn = (long)blockIdx.x * BN;

  v8f acc[2][2];
#pragma unroll
  for (int i = 0; i < 2; ++i)
#pragma unroll
    for (int j = 0; j < 2; ++j) acc[i][j] = vzero8();

  for (int k0 = 0; k0 < K; k0 += BK) {
    if (k0 + BK < K) {  // prefetch next K tile (global_prefetch_b8)
      __builtin_prefetch(A + (bm + (tid >> 1)) * (long)K + k0 + BK, 0, 1);
      __builtin_prefetch(B + (long)(k0 + BK + (tid >> 3)) * N + bn, 0, 1);
    }
    // A tile: 128x64 f32 -> bf16, packed pair stores (4096 b32 stores / block)
#pragma unroll
    for (int i = 0; i < 16; ++i) {
      int idx = tid + i * 256;              // uint-pair index
      int r = idx >> 5, c2 = idx & 31;      // 32 uints per row
      v2f f = *(const v2f*)(A + (bm + r) * (long)K + k0 + 2 * c2);
      *(unsigned int*)&As[r][2 * c2] = f2bf2(f.x, f.y);
    }
    // B tile: 64x64 f32, stored transposed Bt[n][k]
#pragma unroll
    for (int i = 0; i < 8; ++i) {
      int idx = tid + i * 256;
      int r = idx >> 5, c2 = idx & 31;      // k = r, n = 2*c2(+1)
      v2f f = *(const v2f*)(B + (long)(k0 + r) * N + bn + 2 * c2);
      Bt[2 * c2][r]     = f2bf(f.x);
      Bt[2 * c2 + 1][r] = f2bf(f.y);
    }
    __syncthreads();

#pragma unroll
    for (int ks = 0; ks < BK; ks += 32) {
      v16bf af[2], bfr[2];
#pragma unroll
      for (int i = 0; i < 2; ++i) {         // A frags: 2x ds_load_b128 each
        frag_u t;
        int mr = wm * 32 + i * 16 + mlo;
        t.u4[0] = *(const v4u*)&As[mr][ks + kb8];
        t.u4[1] = *(const v4u*)&As[mr][ks + kb8 + 16];
        af[i] = __builtin_bit_cast(v16bf, t.u8);
      }
#pragma unroll
      for (int j = 0; j < 2; ++j) {         // B frags: 2x ds_load_b128 each
        frag_u t;
        int nc = wn * 32 + j * 16 + mlo;
        t.u4[0] = *(const v4u*)&Bt[nc][ks + kb8];
        t.u4[1] = *(const v4u*)&Bt[nc][ks + kb8 + 16];
        bfr[j] = __builtin_bit_cast(v16bf, t.u8);
      }
#pragma unroll
      for (int i = 0; i < 2; ++i)
#pragma unroll
        for (int j = 0; j < 2; ++j)
          acc[i][j] = wmma_bf16(af[i], bfr[j], acc[i][j]);
    }
    __syncthreads();
  }

  const long cbase = (bm + wm * 32 + 8 * hi) * (long)N + bn + wn * 32 + mlo;
#pragma unroll
  for (int i = 0; i < 2; ++i)
#pragma unroll
    for (int j = 0; j < 2; ++j) {
      float* p = C + cbase + (long)i * 16 * N + j * 16;
#pragma unroll
      for (int r = 0; r < 8; ++r) { *p = acc[i][j][r]; p += N; }
    }
}

// ---------------------------------------------------------------------------
// RMSNorm: one block per row. Supports strided sub-view input.
// ---------------------------------------------------------------------------
__global__ __launch_bounds__(256) void rmsnorm_kernel(
    const float* __restrict__ x, const float* __restrict__ w,
    float* __restrict__ y, int n, int in_stride, int out_stride) {
  const long row = blockIdx.x;
  const float* xr = x + row * (long)in_stride;
  float* yr = y + row * (long)out_stride;
  float s = 0.0f;
  for (int i = threadIdx.x; i < n; i += 256) { float v = xr[i]; s += v * v; }
  __shared__ float red[256];
  red[threadIdx.x] = s;
  __syncthreads();
  for (int off = 128; off > 0; off >>= 1) {
    if (threadIdx.x < off) red[threadIdx.x] += red[threadIdx.x + off];
    __syncthreads();
  }
  float scale = rsqrtf(red[0] / (float)n + 1e-6f);
  for (int i = threadIdx.x; i < n; i += 256) yr[i] = xr[i] * scale * w[i];
}

// ---------------------------------------------------------------------------
// Pack kernels: fuse RoPE + fp32->bf16, build contiguous Q/K/V for flash.
// blockIdx.x = s*NH + h
// ---------------------------------------------------------------------------
__global__ __launch_bounds__(96) void pack_q_kernel(
    const float* __restrict__ q_raw, const int* __restrict__ positions,
    unsigned short* __restrict__ Qp) {
  const int s = blockIdx.x / NH, h = blockIdx.x % NH;
  const float* src = q_raw + ((long)s * NH + h) * QKD;
  unsigned short* dst = Qp + ((long)s * NH + h) * QKD;
  const int d2 = threadIdx.x;
  if (d2 < NOPE / 2) {
    v2f f = *(const v2f*)(src + 2 * d2);
    *(unsigned int*)(dst + 2 * d2) = f2bf2(f.x, f.y);
  } else {
    const int i = d2 - NOPE / 2;           // 0..31
    float pos = (float)positions[s];
    float ang = pos * __powf(10000.0f, -(float)i / 32.0f);
    float sn, cs; sincosf(ang, &sn, &cs);
    float x1 = src[NOPE + 2 * i], x2 = src[NOPE + 2 * i + 1];
    *(unsigned int*)(dst + NOPE + 2 * i) =
        f2bf2(x1 * cs - x2 * sn, x2 * cs + x1 * sn);
  }
}

__global__ __launch_bounds__(96) void pack_k_kernel(
    const float* __restrict__ kvmat, const float* __restrict__ kv_c,
    const int* __restrict__ positions, unsigned short* __restrict__ Kp) {
  const int s = blockIdx.x / NH, h = blockIdx.x % NH;
  unsigned short* dst = Kp + ((long)s * NH + h) * QKD;
  const int d2 = threadIdx.x;
  if (d2 < NOPE / 2) {
    const float* src = kvmat + (long)s * (NH * KV_OUT) + h * KV_OUT;
    v2f f = *(const v2f*)(src + 2 * d2);
    *(unsigned int*)(dst + 2 * d2) = f2bf2(f.x, f.y);
  } else {
    const int i = d2 - NOPE / 2;
    const float* src = kv_c + (long)s * (KV_LORA + ROPE) + KV_LORA;
    float pos = (float)positions[s];
    float ang = pos * __powf(10000.0f, -(float)i / 32.0f);
    float sn, cs; sincosf(ang, &sn, &cs);
    float x1 = src[2 * i], x2 = src[2 * i + 1];
    *(unsigned int*)(dst + NOPE + 2 * i) =
        f2bf2(x1 * cs - x2 * sn, x2 * cs + x1 * sn);
  }
}

__global__ __launch_bounds__(128) void pack_v_kernel(
    const float* __restrict__ kvmat, unsigned short* __restrict__ Vp) {
  const int s = blockIdx.x / NH, h = blockIdx.x % NH;
  const int d2 = threadIdx.x;                // 0..63 used twice: 128 elems
  v2f f = *(const v2f*)(kvmat + (long)s * (NH * KV_OUT) + h * KV_OUT + NOPE + 2 * (d2 & 63));
  if (d2 < 64)
    *(unsigned int*)(Vp + ((long)s * NH + h) * V_DIM + 2 * d2) = f2bf2(f.x, f.y);
}

// ---------------------------------------------------------------------------
// Flash attention: block = (64 query rows, 1 head), 4 waves, wave owns 16 rows.
// 32-key tiles, online softmax; QK^T and PV via bf16 WMMA; V staged transposed
// so PV B-fragments are ds_load_b128.
// ---------------------------------------------------------------------------
__global__ __launch_bounds__(128) void flash_kernel(
    const unsigned short* __restrict__ Qp, const unsigned short* __restrict__ Kp,
    const unsigned short* __restrict__ Vp, float* __restrict__ attn) {
  __shared__ unsigned short Ks[32][QKD + 8];   // 32 keys x 192 (stride 400 B)
  __shared__ unsigned short Vst[V_DIM][40];    // transposed: Vst[d][key]
  __shared__ unsigned short Ps[4][16][40];     // per-wave P re-layout bounce
  const int h    = blockIdx.y;
  const int qblk = blockIdx.x;                 // 64-row query block
  const int tid  = threadIdx.x;
  const int wave = tid >> 5, lane = tid & 31;
  const int mlo = lane & 15, hi = lane >> 4;
  const int kb8 = hi * 8;
  const int qbase = qblk * 64 + wave * 16;

  // Q fragments: 6 x (16x32 bf16 A-layout), two global b128 loads each
  v16bf qf[6];
  {
    const unsigned short* qrow = Qp + ((long)(qbase + mlo) * NH + h) * QKD;
#pragma unroll
    for (int f = 0; f < 6; ++f) {
      frag_u t;
      t.u4[0] = *(const v4u*)(qrow + 32 * f + kb8);
      t.u4[1] = *(const v4u*)(qrow + 32 * f + kb8 + 16);
      qf[f] = __builtin_bit_cast(v16bf, t.u8);
    }
  }

  v8f acc[8];
  float m_i[8], l_i[8];
#pragma unroll
  for (int r = 0; r < 8; ++r) { m_i[r] = -1e30f; l_i[r] = 0.0f; }
#pragma unroll
  for (int c = 0; c < 8; ++c) acc[c] = vzero8();

  const float scaling = 0.0721687836f;         // 192^-0.5
  const int nt = (qblk + 1) * 2;               // causal
  for (int t = 0; t < nt; ++t) {
    __syncthreads();                           // previous tile fully consumed
    {
      const unsigned short* kb = Kp + ((long)(t * 32) * NH + h) * QKD;
#pragma unroll
      for (int i = 0; i < 24; ++i) {           // K: 3072 dwords
        int idx = tid + i * 128;
        int r = idx / 96, c2 = idx % 96;
        *(unsigned int*)&Ks[r][2 * c2] =
            *(const unsigned int*)(kb + (long)r * NH * QKD + 2 * c2);
      }
      const unsigned short* vb = Vp + ((long)(t * 32) * NH + h) * V_DIM;
#pragma unroll
      for (int i = 0; i < 16; ++i) {           // V: 2048 dwords, transposed
        int idx = tid + i * 128;
        int key = idx >> 6, d2 = idx & 63;
        unsigned int u =
            *(const unsigned int*)(vb + (long)key * NH * V_DIM + 2 * d2);
        Vst[2 * d2][key]     = (unsigned short)u;
        Vst[2 * d2 + 1][key] = (unsigned short)(u >> 16);
      }
    }
    __syncthreads();

    // S = Q * K^T : two 16x16 output tiles (keys n, n+16)
    v8f c0 = vzero8(), c1 = vzero8();
#pragma unroll
    for (int f = 0; f < 6; ++f) {
      frag_u t0, t1;
      t0.u4[0] = *(const v4u*)&Ks[mlo][32 * f + kb8];
      t0.u4[1] = *(const v4u*)&Ks[mlo][32 * f + kb8 + 16];
      t1.u4[0] = *(const v4u*)&Ks[16 + mlo][32 * f + kb8];
      t1.u4[1] = *(const v4u*)&Ks[16 + mlo][32 * f + kb8 + 16];
      c0 = wmma_bf16(qf[f], __builtin_bit_cast(v16bf, t0.u8), c0);
      c1 = wmma_bf16(qf[f], __builtin_bit_cast(v16bf, t1.u8), c1);
    }

    // online softmax per row (row = qbase + r + 8*hi; lane column = mlo)
#pragma unroll
    for (int r = 0; r < 8; ++r) {
      int row  = qbase + r + 8 * hi;
      int col0 = t * 32 + mlo;
      float s0 = c0[r] * scaling; if (col0 > row)      s0 = -1e30f;
      float s1 = c1[r] * scaling; if (col0 + 16 > row) s1 = -1e30f;
      float mx = fmaxf(s0, s1);
#pragma unroll
      for (int o = 1; o < 16; o <<= 1) mx = fmaxf(mx, __shfl_xor(mx, o, 32));
      float mnew = fmaxf(m_i[r], mx);
      float corr = __expf(m_i[r] - mnew);
      float p0 = __expf(s0 - mnew);
      float p1 = __expf(s1 - mnew);
      float rs = p0 + p1;
#pragma unroll
      for (int o = 1; o < 16; o <<= 1) rs += __shfl_xor(rs, o, 32);
      l_i[r] = l_i[r] * corr + rs;
      m_i[r] = mnew;
#pragma unroll
      for (int ch = 0; ch < 8; ++ch) acc[ch][r] *= corr;
      int prow = r + 8 * hi;
      Ps[wave][prow][mlo]      = f2bf(p0);
      Ps[wave][prow][16 + mlo] = f2bf(p1);
    }
    asm volatile("s_wait_dscnt 0" ::: "memory"); // intra-wave LDS visibility

    // P (16x32) as A fragment, then O += P * V
    v16bf pf;
    {
      frag_u tp;
      const unsigned short* pr = &Ps[wave][mlo][0];
      tp.u4[0] = *(const v4u*)(pr + kb8);
      tp.u4[1] = *(const v4u*)(pr + kb8 + 16);
      pf = __builtin_bit_cast(v16bf, tp.u8);
    }
#pragma unroll
    for (int ch = 0; ch < 8; ++ch) {
      frag_u tv;
      const unsigned short* vr = &Vst[ch * 16 + mlo][0];
      tv.u4[0] = *(const v4u*)(vr + kb8);
      tv.u4[1] = *(const v4u*)(vr + kb8 + 16);
      acc[ch] = wmma_bf16(pf, __builtin_bit_cast(v16bf, tv.u8), acc[ch]);
    }
  }

  // epilogue: normalize, write attn_out (S x H*V_DIM) f32
#pragma unroll
  for (int r = 0; r < 8; ++r) {
    float inv = 1.0f / l_i[r];
    long row = qbase + r + 8 * hi;
    float* orow = attn + row * (long)(NH * V_DIM) + h * V_DIM;
#pragma unroll
    for (int ch = 0; ch < 8; ++ch) orow[ch * 16 + mlo] = acc[ch][r] * inv;
  }
}

// ---------------------------------------------------------------------------
extern "C" void kernel_launch(void* const* d_in, const int* in_sizes, int n_in,
                              void* d_out, int out_size, void* d_ws, size_t ws_size,
                              hipStream_t stream) {
  const float* hidden    = (const float*)d_in[0];
  const float* w_qa      = (const float*)d_in[1];
  const float* qa_ln_w   = (const float*)d_in[2];
  const float* w_qb      = (const float*)d_in[3];
  const float* w_kva     = (const float*)d_in[4];
  const float* kva_ln_w  = (const float*)d_in[5];
  const float* w_kvb     = (const float*)d_in[6];
  const float* w_o       = (const float*)d_in[7];
  const int*   positions = (const int*)d_in[8];
  float* out = (float*)d_out;

  char* w = (char*)d_ws;
  const size_t MB = 1ull << 20;
  float* q_c    = (float*)(w + 0);        // 2048x1536 f32 = 12 MiB
  float* q_raw  = (float*)(w + 12 * MB);  // 2048x3072 f32 = 24 MiB
  float* kv_c   = (float*)(w + 36 * MB);  // 2048x576  f32 = 4.5 MiB
  float* kv_lat = (float*)(w + 41 * MB);  // 2048x512  f32 = 4 MiB
  float* kv     = (float*)(w + 45 * MB);  // 2048x4096 f32 = 32 MiB
  unsigned short* Qp = (unsigned short*)(w + 77 * MB);   // 12 MiB bf16
  unsigned short* Kp = (unsigned short*)(w + 89 * MB);   // 12 MiB bf16
  unsigned short* Vp = (unsigned short*)(w + 101 * MB);  // 8 MiB bf16
  float* attn  = (float*)(w + 12 * MB);   // reuse q_raw slot (dead after pack_q)

  // q path
  gemm_bf16_kernel<<<dim3(Q_LORA / BN, S_LEN / BM), 256, 0, stream>>>(
      hidden, w_qa, q_c, S_LEN, Q_LORA, HIDDEN);
  rmsnorm_kernel<<<S_LEN, 256, 0, stream>>>(q_c, qa_ln_w, q_c, Q_LORA, Q_LORA, Q_LORA);
  gemm_bf16_kernel<<<dim3((NH * QKD) / BN, S_LEN / BM), 256, 0, stream>>>(
      q_c, w_qb, q_raw, S_LEN, NH * QKD, Q_LORA);

  // kv path
  gemm_bf16_kernel<<<dim3((KV_LORA + ROPE) / BN, S_LEN / BM), 256, 0, stream>>>(
      hidden, w_kva, kv_c, S_LEN, KV_LORA + ROPE, HIDDEN);
  rmsnorm_kernel<<<S_LEN, 256, 0, stream>>>(kv_c, kva_ln_w, kv_lat, KV_LORA,
                                            KV_LORA + ROPE, KV_LORA);
  gemm_bf16_kernel<<<dim3((NH * KV_OUT) / BN, S_LEN / BM), 256, 0, stream>>>(
      kv_lat, w_kvb, kv, S_LEN, NH * KV_OUT, KV_LORA);

  // pack (RoPE fused) -> bf16 Q/K/V
  pack_q_kernel<<<S_LEN * NH, 96, 0, stream>>>(q_raw, positions, Qp);
  pack_k_kernel<<<S_LEN * NH, 96, 0, stream>>>(kv, kv_c, positions, Kp);
  pack_v_kernel<<<S_LEN * NH, 128, 0, stream>>>(kv, Vp);

  // flash attention
  flash_kernel<<<dim3(S_LEN / 64, NH), 128, 0, stream>>>(Qp, Kp, Vp, attn);

  // output projection
  gemm_bf16_kernel<<<dim3(HIDDEN / BN, S_LEN / BM), 256, 0, stream>>>(
      attn, w_o, out, S_LEN, HIDDEN, NH * V_DIM);
}
